// Manifold_69638599737821
// MI455X (gfx1250) — matrix operand order repaired and checked by reading, not compiled, not run on hardware
//
#include <hip/hip_runtime.h>
#include <hip/hip_bf16.h>

// ---------------------------------------------------------------------------
// Manifold regularizer for MI455X (gfx1250, wave32, WMMA, async-to-LDS).
//   Pass 1: split X into bf16 hi/lo (error ~2^-16 rel) + row norms sq[i].
//   Pass 2: G = X X^T via 3 bf16 WMMAs (hi*hi + hi*lo + lo*hi), tiles staged
//           with gfx1250 GLOBAL_LOAD_ASYNC_TO_LDS_B128 (double-buffered),
//           fused running top-2 per row (never materialize 4096^2 matrix),
//           scan parallelized over all 256 threads (4 col-classes per row).
//   Pass 3: only the 8192 selected pairs get ||y_i - y_j|| (1000 dims).
// Workspace use: ~16.6 MB (Xhi/Xlo bf16 copies + candidates) — L2 resident.
// ---------------------------------------------------------------------------

typedef __attribute__((ext_vector_type(16))) __bf16 v16bf;
typedef __attribute__((ext_vector_type(8)))  float  v8f;
typedef __attribute__((ext_vector_type(4)))  float  v4f;
typedef __attribute__((ext_vector_type(4)))  int    v4i;

typedef __attribute__((address_space(1))) v4i g_v4i;  // global int4
typedef __attribute__((address_space(3))) v4i l_v4i;  // LDS int4

#define N_PTS   4096
#define D_FEAT  1024
#define N_CLS   1000
#define ALPHA_C 5.0e-4f

#define TILE_M 64
#define TILE_N 64
#define KSTEP  32
#define KSTEPS (D_FEAT / KSTEP)
#define ASTR   40   // bf16 tile row stride (80B rows: 16B aligned, conflict-free b128)
#define GSTR   68   // f32 C-tile row stride: conflict-free scatter
#define NSPLIT 4

#if defined(__has_builtin)
#if __has_builtin(__builtin_amdgcn_global_load_async_to_lds_b128)
#define HAVE_ASYNC 1
#else
#define HAVE_ASYNC 0
#endif
#else
#define HAVE_ASYNC 0
#endif

__device__ __forceinline__ float kinf() { return __builtin_inff(); }

__device__ __forceinline__ void copy16_to_lds(void* lds_dst, const void* g_src) {
#if HAVE_ASYNC
    __builtin_amdgcn_global_load_async_to_lds_b128(
        (g_v4i*)g_src, (l_v4i*)lds_dst, /*offset=*/0, /*cpol=*/0);
#else
    *(v4f*)lds_dst = *(const v4f*)g_src;
#endif
}

__device__ __forceinline__ void wait_async_le4() {
#if HAVE_ASYNC
#if __has_builtin(__builtin_amdgcn_s_wait_asynccnt)
    __builtin_amdgcn_s_wait_asynccnt(4);
#else
    asm volatile("s_wait_asynccnt 0x4" ::: "memory");
#endif
#endif
}
__device__ __forceinline__ void wait_async_le0() {
#if HAVE_ASYNC
#if __has_builtin(__builtin_amdgcn_s_wait_asynccnt)
    __builtin_amdgcn_s_wait_asynccnt(0);
#else
    asm volatile("s_wait_asynccnt 0x0" ::: "memory");
#endif
#endif
}

// A fragment (16x32 bf16, MxK): lane m<16 -> row m, K {kh..kh+7} U {16+kh..}.
__device__ __forceinline__ v16bf load_fragA(const __bf16* base, int lane) {
    int r  = lane & 15;
    int kh = (lane >> 4) << 3;
    const __bf16* p = base + r * ASTR;
    union { v4f f[2]; v16bf b; } u;
    u.f[0] = *(const v4f*)(p + kh);
    u.f[1] = *(const v4f*)(p + 16 + kh);
    return u.b;
}

// B fragment (32x16 bf16, KxN): lane n<16 -> col n, K {kh..kh+7} U {16+kh..} (kh=0/8).
__device__ __forceinline__ v16bf load_fragB(const __bf16* base, int lane) {
    int n  = lane & 15;
    int kb = (lane >> 4) << 3;
    const __bf16* p = base + n * ASTR + kb;
    union { v4f f[2]; v16bf b; } u;
    u.f[0] = *(const v4f*)(p);
    u.f[1] = *(const v4f*)(p + 16);
    return u.b;
}

__device__ __forceinline__ v8f wmma_bf16(v16bf a, v16bf b, v8f c) {
    return __builtin_amdgcn_wmma_f32_16x16x32_bf16(
        false, a, false, b, (short)0, c, false, false);
}

// ---------------------------------------------------------------------------
// Kernel 1: one block per row. Split X row into bf16 hi/lo, compute sq[row].
// ---------------------------------------------------------------------------
__global__ __launch_bounds__(256) void prep_kernel(const float* __restrict__ X,
                                                   __bf16* __restrict__ Xhi,
                                                   __bf16* __restrict__ Xlo,
                                                   float* __restrict__ sq,
                                                   float* __restrict__ out) {
    __shared__ float part[8];
    const int tid  = threadIdx.x;
    const int lane = tid & 31;
    const int wid  = tid >> 5;
    const int row  = blockIdx.x;
    const size_t rb = (size_t)row * D_FEAT;
    const int c4 = tid * 4;

    v4f x = *(const v4f*)(X + rb + c4);
    union { __bf16 b[4]; uint2 u; } uh, ul;
    float s = 0.f;
    #pragma unroll
    for (int k = 0; k < 4; ++k) {
        float v = x[k];
        s += v * v;
        __bf16 h = (__bf16)v;
        uh.b[k] = h;
        ul.b[k] = (__bf16)(v - (float)h);
    }
    *(uint2*)(Xhi + rb + c4) = uh.u;
    *(uint2*)(Xlo + rb + c4) = ul.u;

    for (int o = 16; o; o >>= 1) s += __shfl_xor(s, o, 32);
    if (lane == 0) part[wid] = s;
    __syncthreads();
    if (tid == 0) {
        float t = 0.f;
        #pragma unroll
        for (int w = 0; w < 8; ++w) t += part[w];
        sq[row] = t;
        if (blockIdx.x == 0) out[0] = 0.f;
    }
}

// ---------------------------------------------------------------------------
// Kernel 2: fused GEMM (bf16 hi/lo split) + per-row top-2.
// Grid: 64 row-blocks x 4 column-splits; 8 waves; wave = 16x32 strip.
// Thread (row=tid&63, qr=tid>>6) owns column-class [16*qr,16*qr+16) of every
// tile and keeps a running top-2 in registers; classes merged once at the end.
// ---------------------------------------------------------------------------
__global__ __launch_bounds__(256) void knn_gemm_kernel(const __bf16* __restrict__ Xhi,
                                                       const __bf16* __restrict__ Xlo,
                                                       const float* __restrict__ sq,
                                                       float* __restrict__ cand_d2,
                                                       int*   __restrict__ cand_idx) {
    __shared__ __bf16 tiles[2][4][TILE_M * ASTR];   // [buf][Ahi,Alo,Bhi,Blo]
    __shared__ float  Gt[TILE_M * GSTR];
    __shared__ float  sqj_s[TILE_N];
    __shared__ float  csd[512];                     // 64 rows x 4 classes x 2
    __shared__ int    csi[512];

    const int tid  = threadIdx.x;
    const int lane = tid & 31;
    const int wid  = tid >> 5;
    const int wr   = wid & 3;
    const int wc   = wid >> 2;
    const int rb   = blockIdx.x >> 2;
    const int js   = blockIdx.x & (NSPLIT - 1);
    const int i_base = rb * TILE_M;

    // staging slot: row sr, 8-elem part sp
    const int sr   = tid >> 2;
    const int sp   = (tid & 3) << 3;
    const int soff = sr * ASTR + sp;

    // scan identity: row + column class
    const int srow = tid & 63;
    const int qr   = tid >> 6;                 // 0..3 -> cols [16*qr, 16*qr+16)
    const float sqi  = sq[i_base + srow];
    const int   irow = i_base + srow;

    float bd0 = kinf(), bd1 = kinf();
    int   bj0 = -1,     bj1 = -1;

    for (int jt = 0; jt < (N_PTS / NSPLIT) / TILE_N; ++jt) {
        const int j_base = js * (N_PTS / NSPLIT) + jt * TILE_N;
        v8f acc0 = {}; v8f acc1 = {};
        if (tid < TILE_N) sqj_s[tid] = sq[j_base + tid];

        const size_t abase = (size_t)(i_base + sr) * D_FEAT + sp;
        const size_t bbase = (size_t)(j_base + sr) * D_FEAT + sp;

        copy16_to_lds(&tiles[0][0][soff], Xhi + abase);
        copy16_to_lds(&tiles[0][1][soff], Xlo + abase);
        copy16_to_lds(&tiles[0][2][soff], Xhi + bbase);
        copy16_to_lds(&tiles[0][3][soff], Xlo + bbase);

        for (int s = 0; s < KSTEPS; ++s) {
            const int cur = s & 1;
            if (s + 1 < KSTEPS) {
                const int nxt = cur ^ 1;
                const size_t ko = (size_t)(s + 1) * KSTEP;
                copy16_to_lds(&tiles[nxt][0][soff], Xhi + abase + ko);
                copy16_to_lds(&tiles[nxt][1][soff], Xlo + abase + ko);
                copy16_to_lds(&tiles[nxt][2][soff], Xhi + bbase + ko);
                copy16_to_lds(&tiles[nxt][3][soff], Xlo + bbase + ko);
                wait_async_le4();
            } else {
                wait_async_le0();
            }
            __syncthreads();

            v16bf ah  = load_fragA(&tiles[cur][0][(16 * wr) * ASTR], lane);
            v16bf al  = load_fragA(&tiles[cur][1][(16 * wr) * ASTR], lane);
            v16bf b0h = load_fragB(&tiles[cur][2][(32 * wc) * ASTR], lane);
            v16bf b0l = load_fragB(&tiles[cur][3][(32 * wc) * ASTR], lane);
            v16bf b1h = load_fragB(&tiles[cur][2][(32 * wc + 16) * ASTR], lane);
            v16bf b1l = load_fragB(&tiles[cur][3][(32 * wc + 16) * ASTR], lane);

            acc0 = wmma_bf16(ah, b0h, acc0);
            acc0 = wmma_bf16(ah, b0l, acc0);
            acc0 = wmma_bf16(al, b0h, acc0);
            acc1 = wmma_bf16(ah, b1h, acc1);
            acc1 = wmma_bf16(ah, b1l, acc1);
            acc1 = wmma_bf16(al, b1h, acc1);

            __syncthreads();
        }

        // Spill C tiles (lanes 0-15: N=lane, M=v; lanes 16-31: N=lane-16, M=v+8)
        {
            int n  = lane & 15;
            int mo = (lane >> 4) << 3;
            #pragma unroll
            for (int v = 0; v < 8; ++v) {
                int m = 16 * wr + mo + v;
                Gt[m * GSTR + 32 * wc + n]      = acc0[v];
                Gt[m * GSTR + 32 * wc + 16 + n] = acc1[v];
            }
        }
        __syncthreads();

        // Parallel running top-2: every thread scans its 16-column class.
        {
            const float* grow = Gt + srow * GSTR + 16 * qr;
            const float* sjq  = sqj_s + 16 * qr;
            #pragma unroll 4
            for (int c = 0; c < 16; ++c) {
                int   j  = j_base + 16 * qr + c;
                float d2 = fmaxf(sqi + sjq[c] - 2.0f * grow[c], 0.0f);
                if (j == irow) d2 = kinf();
                if (d2 < bd0)      { bd1 = bd0; bj1 = bj0; bd0 = d2; bj0 = j; }
                else if (d2 < bd1) { bd1 = d2; bj1 = j; }
            }
        }
        __syncthreads();
    }

    // Merge the 4 column-classes per row (8 candidates) -> split top-2.
    csd[srow * 8 + qr * 2]     = bd0;  csi[srow * 8 + qr * 2]     = bj0;
    csd[srow * 8 + qr * 2 + 1] = bd1;  csi[srow * 8 + qr * 2 + 1] = bj1;
    __syncthreads();
    if (tid < TILE_M) {
        const float* cd = csd + tid * 8;
        const int*   ci = csi + tid * 8;
        float d0 = kinf(); int j0 = 0x7fffffff, s0 = -1;
        #pragma unroll
        for (int s = 0; s < 8; ++s) {
            float dd = cd[s]; int jj = ci[s];
            if (dd < d0 || (dd == d0 && jj < j0)) { d0 = dd; j0 = jj; s0 = s; }
        }
        float d1 = kinf(); int j1 = 0x7fffffff;
        #pragma unroll
        for (int s = 0; s < 8; ++s) {
            if (s == s0) continue;
            float dd = cd[s]; int jj = ci[s];
            if (dd < d1 || (dd == d1 && jj < j1)) { d1 = dd; j1 = jj; }
        }
        int o = (i_base + tid) * (2 * NSPLIT) + js * 2;
        cand_d2[o]     = d0;  cand_d2[o + 1]  = d1;
        cand_idx[o]    = j0;  cand_idx[o + 1] = j1;
    }
}

// ---------------------------------------------------------------------------
// Kernel 3: one wave per (i,k) pair: merge split candidates, ||y_i-y_j||,
// accumulate ALPHA * sign * exp(-d) * ||dy||.
// ---------------------------------------------------------------------------
__global__ __launch_bounds__(256) void pairs_kernel(const float* __restrict__ cand_d2,
                                                    const int*   __restrict__ cand_idx,
                                                    const int*   __restrict__ yb,
                                                    const float* __restrict__ Y,
                                                    float* __restrict__ out) {
    int lane = threadIdx.x & 31;
    int wid  = threadIdx.x >> 5;
    int p    = blockIdx.x * 8 + wid;
    int i    = p >> 1;
    int kk   = p & 1;

    const float* cd = cand_d2 + i * (2 * NSPLIT);
    const int*   ci = cand_idx + i * (2 * NSPLIT);

    float d0 = kinf(); int j0 = 0x7fffffff, s0 = -1;
    #pragma unroll
    for (int s = 0; s < 2 * NSPLIT; ++s) {
        float dd = cd[s]; int jj = ci[s];
        if (dd < d0 || (dd == d0 && jj < j0)) { d0 = dd; j0 = jj; s0 = s; }
    }
    float d1 = kinf(); int j1 = 0x7fffffff;
    #pragma unroll
    for (int s = 0; s < 2 * NSPLIT; ++s) {
        if (s == s0) continue;
        float dd = cd[s]; int jj = ci[s];
        if (dd < d1 || (dd == d1 && jj < j1)) { d1 = dd; j1 = jj; }
    }
    float d2v = kk ? d1 : d0;
    int   j   = kk ? j1 : j0;

    const float* yi = Y + (size_t)i * N_CLS;
    const float* yj = Y + (size_t)j * N_CLS;
    float s = 0.f;
    for (int c = lane; c < N_CLS; c += 32) {
        float df = yi[c] - yj[c];
        s += df * df;
    }
    for (int o = 16; o; o >>= 1) s += __shfl_xor(s, o, 32);

    if (lane == 0) {
        float ynorm = (s   > 0.f) ? sqrtf(s)   : 0.f;
        float dist  = (d2v > 0.f) ? sqrtf(d2v) : 0.f;
        float sgn   = (yb[i] == yb[j]) ? 1.f : -1.f;
        atomicAdd(out, ALPHA_C * sgn * expf(-dist) * ynorm);
    }
}

// ---------------------------------------------------------------------------
extern "C" void kernel_launch(void* const* d_in, const int* in_sizes, int n_in,
                              void* d_out, int out_size, void* d_ws, size_t ws_size,
                              hipStream_t stream) {
    (void)in_sizes; (void)n_in; (void)out_size; (void)ws_size;
    const float* X  = (const float*)d_in[0];
    const int*   yb = (const int*)d_in[1];
    const float* Y  = (const float*)d_in[2];
    float* out = (float*)d_out;

    char* ws = (char*)d_ws;
    float*  sq       = (float*)ws;                            // 4096 f32
    float*  cand_d2  = (float*)(ws + N_PTS * 4);              // 4096*8 f32
    int*    cand_idx = (int*)(ws + N_PTS * 4 + N_PTS * 32);   // 4096*8 i32
    __bf16* Xhi      = (__bf16*)(ws + N_PTS * 4 + N_PTS * 64);
    __bf16* Xlo      = Xhi + (size_t)N_PTS * D_FEAT;

    prep_kernel<<<N_PTS, 256, 0, stream>>>(X, Xhi, Xlo, sq, out);
    knn_gemm_kernel<<<(N_PTS / TILE_M) * NSPLIT, 256, 0, stream>>>(Xhi, Xlo, sq, cand_d2, cand_idx);
    pairs_kernel<<<(N_PTS * 2) / 8, 256, 0, stream>>>(cand_d2, cand_idx, yb, Y, out);
}
